// TransformerRegressor_15736760173140
// MI455X (gfx1250) — matrix operand order repaired and checked by reading, not compiled, not run on hardware
//
#include <hip/hip_runtime.h>

typedef __attribute__((ext_vector_type(16))) _Float16 v16h;
typedef __attribute__((ext_vector_type(8)))  _Float16 v8h;
typedef __attribute__((ext_vector_type(8)))  float    v8f;

union V16 { v16h v; v8h h[2]; };

__device__ __forceinline__ v8f wmma32(v16h a, v16h b, v8f c) {
  // D = A(16x32 f16) * B(32x16 f16) + C(16x16 f32)
  return __builtin_amdgcn_wmma_f32_16x16x32_f16(false, a, false, b, (short)0, c,
                                                false, false);
}

// ---------------------------------------------------------------------------
// Embedding: x[b,l,:] = tok_emb[ids[b,l]] + pos_emb[l];  kpm = 0 or -1e9
// ---------------------------------------------------------------------------
__global__ void embed_kernel(const int* __restrict__ ids,
                             const float* __restrict__ tok,
                             const float* __restrict__ pos,
                             float* __restrict__ x, float* __restrict__ kpm,
                             int Lx, int D) {
  const int l = blockIdx.x, b = blockIdx.y;
  const int id = ids[(size_t)b * Lx + l];
  for (int c = threadIdx.x; c < D; c += blockDim.x)
    x[((size_t)b * Lx + l) * D + c] =
        tok[(size_t)id * D + c] + pos[(size_t)l * D + c];
  if (threadIdx.x == 0)
    kpm[(size_t)b * Lx + l] = (id == 0 && l != 0) ? -1.0e9f : 0.0f;
}

// ---------------------------------------------------------------------------
// GEMM: C[M,N] = act( A[M,K] * W[N,K]^T + bias ), f32 in/out, f16 WMMA core.
// Block tile 128x128, 8 waves, wave tile 32x64. Double-buffered LDS:
// next panel's global loads are in flight under the current panel's WMMAs.
// ---------------------------------------------------------------------------
#define GT_M 128
#define GT_N 128
#define GT_K 32
#define LDH  40   // LDS stride in halves (80B, multiple of 16B)

__global__ __launch_bounds__(256)
void gemm_bias_act(const float* __restrict__ A, const float* __restrict__ W,
                   const float* __restrict__ bias, float* __restrict__ C,
                   int M, int N, int K, int act) {
  __shared__ _Float16 aT[2][GT_M * LDH];
  __shared__ _Float16 bT[2][GT_N * LDH];

  const int tid  = threadIdx.x;
  const int lane = tid & 31;
  const int wave = tid >> 5;
  const int bm = blockIdx.x * GT_M;
  const int bn = blockIdx.y * GT_N;
  const int wm = (wave & 3) * 32;   // wave row offset in block tile
  const int wn = (wave >> 2) * 64;  // wave col offset in block tile

  v8f acc[2][4] = {};

  const int lrow = tid >> 3;        // 0..31 staging row
  const int lcol = (tid & 7) * 4;   // 0..28 staging col (x4 floats)
  const float* Abase = A + (size_t)(bm + lrow) * K + lcol;
  const float* Wbase = W + (size_t)(bn + lrow) * K + lcol;
  const size_t rskip = (size_t)32 * K;  // 32 rows

  // ISA 16-bit fragment lane layout
  const int am  = lane & 15;        // A row / B col within 16x16 tile
  const int ak  = (lane >> 4) * 8;  // A: halves {ak..ak+7, ak+16..ak+23}
  const int bk  = (lane >> 4) * 8;  // B: halves {bk..bk+7, bk+16..bk+23}
  const int hi8 = (lane >> 4) * 8;  // C/D row offset for upper half-wave

  float4 ra[4], rb[4];

  // ---- prologue: stage panel 0 into buffer 0 ----
#pragma unroll
  for (int it = 0; it < 4; ++it) {
    ra[it] = *(const float4*)(Abase + it * rskip);
    rb[it] = *(const float4*)(Wbase + it * rskip);
  }
#pragma unroll
  for (int it = 0; it < 4; ++it) {
    const int row = lrow + it * 32;
    _Float16* da = &aT[0][row * LDH + lcol];
    da[0] = (_Float16)ra[it].x; da[1] = (_Float16)ra[it].y;
    da[2] = (_Float16)ra[it].z; da[3] = (_Float16)ra[it].w;
    _Float16* db = &bT[0][row * LDH + lcol];
    db[0] = (_Float16)rb[it].x; db[1] = (_Float16)rb[it].y;
    db[2] = (_Float16)rb[it].z; db[3] = (_Float16)rb[it].w;
  }
  __syncthreads();

  const int nk = K / GT_K;
  for (int step = 0; step < nk; ++step) {
    const int cur = step & 1;
    const bool has_next = (step + 1) < nk;

    // issue next-panel loads first (consumed only after the WMMA block)
    if (has_next) {
      const int kn = (step + 1) * GT_K;
#pragma unroll
      for (int it = 0; it < 4; ++it) {
        ra[it] = *(const float4*)(Abase + kn + it * rskip);
        rb[it] = *(const float4*)(Wbase + kn + it * rskip);
      }
      if (step + 2 < nk) {  // global_prefetch_b8 two panels ahead
        __builtin_prefetch(Abase + (size_t)(step + 2) * GT_K, 0, 1);
        __builtin_prefetch(Wbase + (size_t)(step + 2) * GT_K, 0, 1);
      }
    }

    // compute on current buffer
    V16 af[2];
#pragma unroll
    for (int i = 0; i < 2; ++i) {
      const _Float16* p = &aT[cur][(wm + i * 16 + am) * LDH + ak];
      af[i].h[0] = *(const v8h*)(p);
      af[i].h[1] = *(const v8h*)(p + 16);
    }
#pragma unroll
    for (int j = 0; j < 4; ++j) {
      const _Float16* p = &bT[cur][(wn + j * 16 + am) * LDH + bk];
      V16 bf;
      bf.h[0] = *(const v8h*)(p);
      bf.h[1] = *(const v8h*)(p + 16);
#pragma unroll
      for (int i = 0; i < 2; ++i)
        acc[i][j] = wmma32(af[i].v, bf.v, acc[i][j]);
    }

    // convert + store next panel into the other buffer
    if (has_next) {
      const int nxt = cur ^ 1;
#pragma unroll
      for (int it = 0; it < 4; ++it) {
        const int row = lrow + it * 32;
        _Float16* da = &aT[nxt][row * LDH + lcol];
        da[0] = (_Float16)ra[it].x; da[1] = (_Float16)ra[it].y;
        da[2] = (_Float16)ra[it].z; da[3] = (_Float16)ra[it].w;
        _Float16* db = &bT[nxt][row * LDH + lcol];
        db[0] = (_Float16)rb[it].x; db[1] = (_Float16)rb[it].y;
        db[2] = (_Float16)rb[it].z; db[3] = (_Float16)rb[it].w;
      }
    }
    __syncthreads();
  }

#pragma unroll
  for (int j = 0; j < 4; ++j) {
    const int col = bn + wn + j * 16 + am;
    const float bv = bias ? bias[col] : 0.0f;
#pragma unroll
    for (int i = 0; i < 2; ++i) {
      const int rowb = bm + wm + i * 16 + hi8;
#pragma unroll
      for (int r = 0; r < 8; ++r) {
        float v = acc[i][j][r] + bv;
        if (act == 1)  // exact GELU
          v = 0.5f * v * (1.0f + erff(v * 0.70710678118f));
        C[(size_t)(rowb + r) * N + col] = v;
      }
    }
  }
}

// ---------------------------------------------------------------------------
// Flash attention: one block per (64-query tile, head, batch). dh = 64.
// 8 waves; wave owns 2 of the 16 (16x16) tiles of the 64x64 S / O blocks.
// K/V (+mask) are double-buffered: next key-block's loads fly under both
// WMMA phases and the softmax of the current block.
// ---------------------------------------------------------------------------
#define AT_S 72  // LDS stride in halves (144B, multiple of 16B)

__global__ __launch_bounds__(256)
void flash_attn(const float* __restrict__ qkv, const float* __restrict__ kpm,
                float* __restrict__ out, int Lx, int D, int H) {
  const int dh = 64;
  const int qt = blockIdx.x, h = blockIdx.y, b = blockIdx.z;
  const int tid = threadIdx.x, lane = tid & 31, wave = tid >> 5;

  __shared__ _Float16 qs[64 * AT_S];      // Q tile  [q][dh], pre-scaled
  __shared__ _Float16 ksh[2][64 * AT_S];  // K tile  [key][dh]
  __shared__ _Float16 vth[2][64 * AT_S];  // V tile^T [dh][key]
  __shared__ _Float16 pbh[64 * AT_S];     // P tile  [q][key] (f16)
  __shared__ float sb[64 * 65];           // S tile  f32
  __shared__ float maskv[2][64];
  __shared__ float corr[64];
  __shared__ float linv[64];

  const size_t rstride = (size_t)3 * D;  // qkv row stride (3072)
  const float* Qg = qkv + ((size_t)b * Lx) * rstride + (size_t)0 * D + h * dh;
  const float* Kg = qkv + ((size_t)b * Lx) * rstride + (size_t)1 * D + h * dh;
  const float* Vg = qkv + ((size_t)b * Lx) * rstride + (size_t)2 * D + h * dh;

  const int sr = tid >> 2;         // staging row 0..63
  const int sc = (tid & 3) * 16;   // staging col base
  const float scale = 0.125f;      // dh^-0.5

  const int tr  = wave >> 1;       // tile row 0..3
  const int tc0 = (wave & 1) * 2;  // first tile col (0 or 2)
  const int am  = lane & 15;
  const int ak  = (lane >> 4) * 8;
  const int bk  = (lane >> 4) * 8;
  const int hi8 = (lane >> 4) * 8;

  // ---- prologue: stage Q and key-block 0 (batched loads, then convert) ----
  float4 rq[4], rk[4], rv[4];
#pragma unroll
  for (int ci = 0; ci < 4; ++ci) {
    rq[ci] = *(const float4*)(Qg + (size_t)(qt * 64 + sr) * rstride + sc + ci * 4);
    rk[ci] = *(const float4*)(Kg + (size_t)sr * rstride + sc + ci * 4);
    rv[ci] = *(const float4*)(Vg + (size_t)sr * rstride + sc + ci * 4);
  }
#pragma unroll
  for (int ci = 0; ci < 4; ++ci) {
    const int c = sc + ci * 4;
    _Float16* dq = &qs[sr * AT_S + c];
    dq[0] = (_Float16)(rq[ci].x * scale); dq[1] = (_Float16)(rq[ci].y * scale);
    dq[2] = (_Float16)(rq[ci].z * scale); dq[3] = (_Float16)(rq[ci].w * scale);
    _Float16* dk = &ksh[0][sr * AT_S + c];
    dk[0] = (_Float16)rk[ci].x; dk[1] = (_Float16)rk[ci].y;
    dk[2] = (_Float16)rk[ci].z; dk[3] = (_Float16)rk[ci].w;
    vth[0][(c + 0) * AT_S + sr] = (_Float16)rv[ci].x;
    vth[0][(c + 1) * AT_S + sr] = (_Float16)rv[ci].y;
    vth[0][(c + 2) * AT_S + sr] = (_Float16)rv[ci].z;
    vth[0][(c + 3) * AT_S + sr] = (_Float16)rv[ci].w;
  }
  if (tid < 64) maskv[0][tid] = kpm[(size_t)b * Lx + tid];
  __syncthreads();

  v8f oacc[2] = {};
  float m_i = -3.0e38f, l_i = 0.0f;  // live in threads 0..63
  float mnext = 0.0f;

  const int nb = Lx / 64;
  for (int kb = 0; kb < nb; ++kb) {
    const int cur = kb & 1;
    const bool hasn = (kb + 1) < nb;

    // issue next key-block loads; consumed only after the P*V WMMAs
    if (hasn) {
#pragma unroll
      for (int ci = 0; ci < 4; ++ci) {
        rk[ci] = *(const float4*)(Kg + (size_t)((kb + 1) * 64 + sr) * rstride + sc + ci * 4);
        rv[ci] = *(const float4*)(Vg + (size_t)((kb + 1) * 64 + sr) * rstride + sc + ci * 4);
      }
      if (tid < 64) mnext = kpm[(size_t)b * Lx + (kb + 1) * 64 + tid];
    }

    // S = Q * K^T  (64x64), 4 WMMA per wave
    v8f sacc[2] = {};
#pragma unroll
    for (int kq = 0; kq < 64; kq += 32) {
      V16 af;
      const _Float16* p = &qs[(tr * 16 + am) * AT_S + kq + ak];
      af.h[0] = *(const v8h*)p; af.h[1] = *(const v8h*)(p + 16);
#pragma unroll
      for (int t = 0; t < 2; ++t) {
        const _Float16* pB = &ksh[cur][((tc0 + t) * 16 + am) * AT_S + kq + bk];
        V16 bf; bf.h[0] = *(const v8h*)pB; bf.h[1] = *(const v8h*)(pB + 16);
        sacc[t] = wmma32(af.v, bf.v, sacc[t]);
      }
    }
    // mask + spill S to LDS
#pragma unroll
    for (int t = 0; t < 2; ++t) {
      const float madd = maskv[cur][(tc0 + t) * 16 + am];
#pragma unroll
      for (int r = 0; r < 8; ++r)
        sb[(tr * 16 + hi8 + r) * 65 + (tc0 + t) * 16 + am] = sacc[t][r] + madd;
    }
    __syncthreads();

    // online softmax (one thread per query row)
    if (tid < 64) {
      float bmx = -3.0e38f;
      for (int c = 0; c < 64; ++c) bmx = fmaxf(bmx, sb[tid * 65 + c]);
      const float mn = fmaxf(m_i, bmx);
      const float cf = __expf(m_i - mn);
      float rs = 0.0f;
      for (int c = 0; c < 64; ++c) {
        const float p = __expf(sb[tid * 65 + c] - mn);
        rs += p;
        pbh[tid * AT_S + c] = (_Float16)p;
      }
      l_i = l_i * cf + rs;
      m_i = mn;
      corr[tid] = cf;
    }
    __syncthreads();

    // rescale O, then O += P * V  (4 WMMA per wave)
    float cf8[8];
#pragma unroll
    for (int r = 0; r < 8; ++r) cf8[r] = corr[tr * 16 + hi8 + r];
#pragma unroll
    for (int t = 0; t < 2; ++t)
#pragma unroll
      for (int r = 0; r < 8; ++r) oacc[t][r] *= cf8[r];

#pragma unroll
    for (int kq = 0; kq < 64; kq += 32) {
      V16 af;
      const _Float16* p = &pbh[(tr * 16 + am) * AT_S + kq + ak];
      af.h[0] = *(const v8h*)p; af.h[1] = *(const v8h*)(p + 16);
#pragma unroll
      for (int t = 0; t < 2; ++t) {
        const _Float16* pB = &vth[cur][((tc0 + t) * 16 + am) * AT_S + kq + bk];
        V16 bf; bf.h[0] = *(const v8h*)pB; bf.h[1] = *(const v8h*)(pB + 16);
        oacc[t] = wmma32(af.v, bf.v, oacc[t]);
      }
    }

    // convert + store next key-block into the other buffer
    if (hasn) {
      const int nxt = cur ^ 1;
#pragma unroll
      for (int ci = 0; ci < 4; ++ci) {
        const int c = sc + ci * 4;
        _Float16* dk = &ksh[nxt][sr * AT_S + c];
        dk[0] = (_Float16)rk[ci].x; dk[1] = (_Float16)rk[ci].y;
        dk[2] = (_Float16)rk[ci].z; dk[3] = (_Float16)rk[ci].w;
        vth[nxt][(c + 0) * AT_S + sr] = (_Float16)rv[ci].x;
        vth[nxt][(c + 1) * AT_S + sr] = (_Float16)rv[ci].y;
        vth[nxt][(c + 2) * AT_S + sr] = (_Float16)rv[ci].z;
        vth[nxt][(c + 3) * AT_S + sr] = (_Float16)rv[ci].w;
      }
      if (tid < 64) maskv[nxt][tid] = mnext;
    }
    __syncthreads();
  }

  if (tid < 64) linv[tid] = 1.0f / l_i;
  __syncthreads();

#pragma unroll
  for (int t = 0; t < 2; ++t) {
    const int col = h * dh + (tc0 + t) * 16 + am;
#pragma unroll
    for (int r = 0; r < 8; ++r) {
      const int q = qt * 64 + tr * 16 + hi8 + r;
      out[((size_t)b * Lx + q) * D + col] =
          oacc[t][r] * linv[tr * 16 + hi8 + r];
    }
  }
}

// ---------------------------------------------------------------------------
// Fused residual add + LayerNorm. One block per row (D == 1024).
// in_stride lets us gather CLS rows (x[:,0,:]) for the head LN.
// ---------------------------------------------------------------------------
__global__ __launch_bounds__(256)
void add_ln(const float* __restrict__ x, const float* __restrict__ res,
            const float* __restrict__ g, const float* __restrict__ bb,
            float* __restrict__ out, int D, size_t in_stride) {
  const int row = blockIdx.x;
  const int tid = threadIdx.x;
  __shared__ float red[256];
  __shared__ float s_mean, s_inv;

  float v4[4];
  float s = 0.0f;
#pragma unroll
  for (int i = 0; i < 4; ++i) {
    const int c = tid + i * 256;
    float v = x[(size_t)row * in_stride + c];
    if (res) v += res[(size_t)row * in_stride + c];
    v4[i] = v; s += v;
  }
  red[tid] = s; __syncthreads();
  for (int st = 128; st > 0; st >>= 1) {
    if (tid < st) red[tid] += red[tid + st];
    __syncthreads();
  }
  if (tid == 0) s_mean = red[0] / (float)D;
  __syncthreads();
  const float mean = s_mean;
  float vs = 0.0f;
#pragma unroll
  for (int i = 0; i < 4; ++i) { const float d = v4[i] - mean; vs += d * d; }
  red[tid] = vs; __syncthreads();
  for (int st = 128; st > 0; st >>= 1) {
    if (tid < st) red[tid] += red[tid + st];
    __syncthreads();
  }
  if (tid == 0) s_inv = rsqrtf(red[0] / (float)D + 1e-5f);
  __syncthreads();
  const float inv = s_inv;
#pragma unroll
  for (int i = 0; i < 4; ++i) {
    const int c = tid + i * 256;
    out[(size_t)row * D + c] = (v4[i] - mean) * inv * g[c] + bb[c];
  }
}

// ---------------------------------------------------------------------------
// Tiny head GEMM: out[b,n] = cls_ln[b,:] . head_W[n,:] + head_b[n]
// ---------------------------------------------------------------------------
__global__ void head_gemm(const float* __restrict__ x, const float* __restrict__ W,
                          const float* __restrict__ bias, float* __restrict__ out,
                          int N, int K) {
  const int n = blockIdx.x * blockDim.x + threadIdx.x;
  const int b = blockIdx.y;
  if (n >= N) return;
  float s = bias[n];
  const float* xr = x + (size_t)b * K;
  const float* wr = W + (size_t)n * K;
  for (int k = 0; k < K; ++k) s += xr[k] * wr[k];
  out[(size_t)b * N + n] = s;
}

// ---------------------------------------------------------------------------
extern "C" void kernel_launch(void* const* d_in, const int* in_sizes, int n_in,
                              void* d_out, int out_size, void* d_ws, size_t ws_size,
                              hipStream_t stream) {
  (void)in_sizes; (void)n_in; (void)out_size; (void)ws_size;

  constexpr int Bc = 8, Lc = 1024, Dc = 1024, Hc = 16, FF = 4096, NL = 6;
  constexpr int M = Bc * Lc;  // 8192 token rows

  const int*   ids    = (const int*)  d_in[0];
  const float* tok    = (const float*)d_in[1];
  const float* pos    = (const float*)d_in[2];
  const float* Wqkv   = (const float*)d_in[3];
  const float* bqkv   = (const float*)d_in[4];
  const float* Wo     = (const float*)d_in[5];
  const float* bo     = (const float*)d_in[6];
  const float* ln1g   = (const float*)d_in[7];
  const float* ln1b   = (const float*)d_in[8];
  const float* W1     = (const float*)d_in[9];
  const float* b1     = (const float*)d_in[10];
  const float* W2     = (const float*)d_in[11];
  const float* b2     = (const float*)d_in[12];
  const float* ln2g   = (const float*)d_in[13];
  const float* ln2b   = (const float*)d_in[14];
  const float* hg     = (const float*)d_in[15];
  const float* hb     = (const float*)d_in[16];
  const float* hW     = (const float*)d_in[17];
  const float* hbias  = (const float*)d_in[18];
  float* outp = (float*)d_out;

  // workspace layout
  float* p    = (float*)d_ws;
  float* x    = p;  p += (size_t)M * Dc;   // 32 MB activations
  float* big  = p;  p += (size_t)M * FF;   // 128 MB: qkv (M x 3072) / ff1 (M x 4096)
  float* att  = p;  p += (size_t)M * Dc;   // 32 MB attention output
  float* tmp  = p;  p += (size_t)M * Dc;   // 32 MB gemm output before LN
  float* kpm  = p;  p += (size_t)M;        // additive mask (0 / -1e9)
  float* cls  = p;  p += (size_t)Bc * Dc;  // LN'ed CLS rows

  embed_kernel<<<dim3(Lc, Bc), 256, 0, stream>>>(ids, tok, pos, x, kpm, Lc, Dc);

  for (int l = 0; l < NL; ++l) {
    const float* lWqkv = Wqkv + (size_t)l * 3 * Dc * Dc;
    const float* lbqkv = bqkv + (size_t)l * 3 * Dc;
    const float* lWo   = Wo   + (size_t)l * Dc * Dc;
    const float* lbo   = bo   + (size_t)l * Dc;
    const float* lW1   = W1   + (size_t)l * FF * Dc;
    const float* lb1   = b1   + (size_t)l * FF;
    const float* lW2   = W2   + (size_t)l * Dc * FF;
    const float* lb2   = b2   + (size_t)l * Dc;

    // QKV projection: [M,1024] x [3072,1024]^T
    gemm_bias_act<<<dim3(M / GT_M, (3 * Dc) / GT_N), 256, 0, stream>>>(
        x, lWqkv, lbqkv, big, M, 3 * Dc, Dc, 0);

    // attention
    flash_attn<<<dim3(Lc / 64, Hc, Bc), 256, 0, stream>>>(big, kpm, att, Lc, Dc, Hc);

    // output projection
    gemm_bias_act<<<dim3(M / GT_M, Dc / GT_N), 256, 0, stream>>>(
        att, lWo, lbo, tmp, M, Dc, Dc, 0);
    // x = LN(x + attn_out)
    add_ln<<<M, 256, 0, stream>>>(tmp, x, ln1g + (size_t)l * Dc,
                                  ln1b + (size_t)l * Dc, x, Dc, (size_t)Dc);

    // FF1 + GELU
    gemm_bias_act<<<dim3(M / GT_M, FF / GT_N), 256, 0, stream>>>(
        x, lW1, lb1, big, M, FF, Dc, 1);
    // FF2
    gemm_bias_act<<<dim3(M / GT_M, Dc / GT_N), 256, 0, stream>>>(
        big, lW2, lb2, tmp, M, Dc, FF, 0);
    // x = LN(x + ff)
    add_ln<<<M, 256, 0, stream>>>(tmp, x, ln2g + (size_t)l * Dc,
                                  ln2b + (size_t)l * Dc, x, Dc, (size_t)Dc);
  }

  // head: LN over CLS rows (x[:,0,:]) then [8,1024] x [1024,1024]^T + bias
  add_ln<<<Bc, 256, 0, stream>>>(x, nullptr, hg, hb, cls, Dc, (size_t)Lc * Dc);
  head_gemm<<<dim3((Lc + 255) / 256, Bc), 256, 0, stream>>>(cls, hW, hbias, outp,
                                                            Lc, Dc);
}
